// TrAISformer_79276506349577
// MI455X (gfx1250) — matrix operand order repaired and checked by reading, not compiled, not run on hardware
//
#include <hip/hip_runtime.h>
#include <hip/hip_bf16.h>
#include <math.h>

// ---------------------------------------------------------------------------
// TrAISformer forward for MI455X (gfx1250, wave32, WMMA f16 -> f32 acc).
// B=64, T=256, D=768, H=12, HD=64, L=12, FULL=302, M = B*T = 16384.
// ---------------------------------------------------------------------------

typedef __attribute__((ext_vector_type(16))) _Float16 v16h;
typedef __attribute__((ext_vector_type(8)))  float    v8f;

#define M_TOK   16384
#define DMODEL  768
#define DFF     3072
#define NLAYER  12
#define NHEAD   12
#define HDIM    64
#define TSEQ    256
#define NFULL   302
#define VSTR    272   // padded stride for transposed V in LDS

// 16 contiguous halves (two 16B chunks) -> v16h
__device__ __forceinline__ v16h load_v16h_contig(const _Float16* p) {
  union { v16h v; uint4 u[2]; } t;
  t.u[0] = *reinterpret_cast<const uint4*>(p);
  t.u[1] = *reinterpret_cast<const uint4*>(p + 8);
  return t.v;
}

// A-operand fragment: lane (lr,lg) holds K = {lg*8..lg*8+7, 16+lg*8..16+lg*8+7}
__device__ __forceinline__ v16h load_a_frag(const _Float16* rowp, int lg) {
  union { v16h v; uint4 u[2]; } t;
  t.u[0] = *reinterpret_cast<const uint4*>(rowp + (lg << 3));
  t.u[1] = *reinterpret_cast<const uint4*>(rowp + 16 + (lg << 3));
  return t.v;
}

// ---------------------------------------------------------------------------
// Embedding: four-hot gather + positional embedding -> h (f32 [M, 768])
// ---------------------------------------------------------------------------
__global__ __launch_bounds__(256) void embed_kernel(
    const int* __restrict__ idxs, const float* __restrict__ lat,
    const float* __restrict__ lon, const float* __restrict__ sog,
    const float* __restrict__ cog, const float* __restrict__ pos,
    float* __restrict__ H) {
  const int tk = blockIdx.x;          // token 0..16383
  const int t  = tk & (TSEQ - 1);     // position in sequence
  const int i0 = idxs[tk * 4 + 0];
  const int i1 = idxs[tk * 4 + 1];
  const int i2 = idxs[tk * 4 + 2];
  const int i3 = idxs[tk * 4 + 3];
  for (int d = threadIdx.x; d < DMODEL; d += 256) {
    float e;
    if (d < 192)       e = lat[i0 * 192 + d];
    else if (d < 384)  e = lon[i1 * 192 + (d - 192)];
    else if (d < 576)  e = sog[i2 * 192 + (d - 384)];
    else               e = cog[i3 * 192 + (d - 576)];
    H[(size_t)tk * DMODEL + d] = e + pos[t * DMODEL + d];
  }
}

// ---------------------------------------------------------------------------
// LayerNorm: f32 row in, f16 row out (GEMM A-operand precision).
// ---------------------------------------------------------------------------
__global__ __launch_bounds__(256) void ln_kernel(
    const float* __restrict__ X, const float* __restrict__ g,
    const float* __restrict__ bta, _Float16* __restrict__ O) {
  const int row = blockIdx.x;
  const int tid = threadIdx.x;
  const float* x = X + (size_t)row * DMODEL;
  float v0 = x[tid], v1 = x[tid + 256], v2 = x[tid + 512];
  float s = v0 + v1 + v2;
  float q = v0 * v0 + v1 * v1 + v2 * v2;
#pragma unroll
  for (int m = 1; m < 32; m <<= 1) {
    s += __shfl_xor(s, m, 32);
    q += __shfl_xor(q, m, 32);
  }
  __shared__ float ss[8], qq[8];
  if ((tid & 31) == 0) { ss[tid >> 5] = s; qq[tid >> 5] = q; }
  __syncthreads();
  float st = 0.f, qt = 0.f;
#pragma unroll
  for (int i = 0; i < 8; ++i) { st += ss[i]; qt += qq[i]; }
  const float mean = st * (1.f / DMODEL);
  const float var  = qt * (1.f / DMODEL) - mean * mean;
  const float rstd = rsqrtf(var + 1e-5f);
  _Float16* o = O + (size_t)row * DMODEL;
  o[tid]       = (_Float16)((v0 - mean) * rstd * g[tid]       + bta[tid]);
  o[tid + 256] = (_Float16)((v1 - mean) * rstd * g[tid + 256] + bta[tid + 256]);
  o[tid + 512] = (_Float16)((v2 - mean) * rstd * g[tid + 512] + bta[tid + 512]);
}

// ---------------------------------------------------------------------------
// Tiled WMMA GEMM:  out = epilogue(A[f16, MxK] * W[f32->f16, KxN] + bias)
// Block: 256 threads (8 waves), tile 128(M) x 64(N), K-step 32.
// Weight tile stored TRANSPOSED in LDS (Bs[col*32+k]) so B-operand
// fragments are two contiguous ds_load_b128 per fragment.
// mode 0: f16 out (QKV)   mode 1: GELU -> f16 (MLP1)
// mode 2: f32 residual += (O-proj, MLP2)   mode 3: f32 out (head)
// ---------------------------------------------------------------------------
__global__ __launch_bounds__(256) void gemm_kernel(
    const _Float16* __restrict__ A, const float* __restrict__ W,
    const float* __restrict__ bias, void* __restrict__ out,
    int K, int N, int mode) {
  __shared__ __align__(16) _Float16 As[128 * 32];
  __shared__ __align__(16) _Float16 Bs[64 * 32];   // [col][k], transposed
  const int tid = threadIdx.x;
  const int bm = blockIdx.y * 128;
  const int bn = blockIdx.x * 64;
  const int w = tid >> 5, lane = tid & 31;
  const int lr = lane & 15, lg = lane >> 4;
  const int wm = (w & 3) * 32, wn = (w >> 2) * 32;

  v8f c[2][2];
#pragma unroll
  for (int i = 0; i < 2; ++i)
#pragma unroll
    for (int j = 0; j < 2; ++j)
#pragma unroll
      for (int r = 0; r < 8; ++r) c[i][j][r] = 0.f;

  const int arow = tid >> 1;
  const int acol = (tid & 1) * 16;
  const int wrow = tid >> 3;         // k within tile, 0..31
  const int wcol = (tid & 7) * 8;    // col base, 0..56

  // uniform: all-of-64-cols valid AND rows 16B aligned (N multiple of 4)
  const bool fastN = ((N & 3) == 0) && (bn + 64 <= N);

  for (int k0 = 0; k0 < K; k0 += 32) {
    // --- stage A tile (f16, 128x32 row-major) ---
    {
      const _Float16* ap = A + (size_t)(bm + arow) * K + k0 + acol;
      const uint4* src = reinterpret_cast<const uint4*>(ap);
      uint4* dst = reinterpret_cast<uint4*>(&As[arow * 32 + acol]);
      dst[0] = src[0];
      dst[1] = src[1];
      __builtin_prefetch(ap + 32, 0, 1);   // next K-step
    }
    // --- stage W tile (f32 -> f16, stored transposed Bs[col][k]) ---
    {
      const float* wsrc = W + (size_t)(k0 + wrow) * N + bn + wcol;
      _Float16* bp = &Bs[wcol * 32 + wrow];
      if (fastN) {
        const float4 f0 = *reinterpret_cast<const float4*>(wsrc);
        const float4 f1 = *reinterpret_cast<const float4*>(wsrc + 4);
        bp[0 * 32] = (_Float16)f0.x;
        bp[1 * 32] = (_Float16)f0.y;
        bp[2 * 32] = (_Float16)f0.z;
        bp[3 * 32] = (_Float16)f0.w;
        bp[4 * 32] = (_Float16)f1.x;
        bp[5 * 32] = (_Float16)f1.y;
        bp[6 * 32] = (_Float16)f1.z;
        bp[7 * 32] = (_Float16)f1.w;
      } else {
        const int vcnt = N - (bn + wcol);
#pragma unroll
        for (int cc = 0; cc < 8; ++cc) {
          float v = (cc < vcnt) ? wsrc[cc] : 0.f;
          bp[cc * 32] = (_Float16)v;
        }
      }
      __builtin_prefetch(wsrc + (size_t)32 * N, 0, 1);  // next K-step
    }
    __syncthreads();

    v16h a[2], bb[2];
#pragma unroll
    for (int i = 0; i < 2; ++i)
      a[i] = load_a_frag(&As[(wm + i * 16 + lr) * 32], lg);
#pragma unroll
    for (int j = 0; j < 2; ++j)
      bb[j] = load_a_frag(&Bs[(wn + j * 16 + lr) * 32], lg);
#pragma unroll
    for (int i = 0; i < 2; ++i)
#pragma unroll
      for (int j = 0; j < 2; ++j)
        c[i][j] = __builtin_amdgcn_wmma_f32_16x16x32_f16(
            false, a[i], false, bb[j], (short)0, c[i][j], false, false);
    __syncthreads();
  }

  // --- epilogue ---
#pragma unroll
  for (int i = 0; i < 2; ++i)
#pragma unroll
    for (int j = 0; j < 2; ++j)
#pragma unroll
      for (int r = 0; r < 8; ++r) {
        const int row = bm + wm + i * 16 + r + 8 * lg;
        const int col = bn + wn + j * 16 + lr;
        if (col >= N) continue;
        float v = c[i][j][r] + (bias ? bias[col] : 0.f);
        const size_t idx = (size_t)row * N + col;
        if (mode == 0) {
          ((_Float16*)out)[idx] = (_Float16)v;
        } else if (mode == 1) {
          const float gl = 0.5f * v * (1.f + erff(v * 0.70710678f));
          ((_Float16*)out)[idx] = (_Float16)gl;
        } else if (mode == 2) {
          ((float*)out)[idx] += v;
        } else {
          ((float*)out)[idx] = v;
        }
      }
}

// ---------------------------------------------------------------------------
// Flash attention, one block per (b, h). 8 waves x 32 query rows = T = 256.
// K (row-major) and V (transposed, padded stride) staged in LDS. Online
// softmax over 16-key tiles; S = Q K^T via 2 WMMAs, P re-laid-out through
// wave-private LDS into the A operand, PV via 4 WMMAs.
// ---------------------------------------------------------------------------
__global__ __launch_bounds__(256) void attn_kernel(
    const _Float16* __restrict__ Qg, const _Float16* __restrict__ Kg,
    const _Float16* __restrict__ Vg, _Float16* __restrict__ Yg) {
  const int h = blockIdx.x;
  const int b = blockIdx.y;
  const int tid = threadIdx.x;
  __shared__ __align__(16) _Float16 Ks[TSEQ * HDIM];    // [key][hd]
  __shared__ __align__(16) _Float16 VsT[HDIM * VSTR];   // [hd][key], padded
  __shared__ __align__(16) _Float16 Ps[8][256];

  {
    const size_t base = ((size_t)(b * TSEQ + tid)) * DMODEL + h * HDIM;
    const uint4* ksrc = reinterpret_cast<const uint4*>(Kg + base);
    const uint4* vsrc = reinterpret_cast<const uint4*>(Vg + base);
    uint4* kdst = reinterpret_cast<uint4*>(&Ks[tid * HDIM]);
    union { uint4 u[8]; _Float16 hh[64]; } vr;
#pragma unroll
    for (int i = 0; i < 8; ++i) { kdst[i] = ksrc[i]; vr.u[i] = vsrc[i]; }
#pragma unroll
    for (int hd = 0; hd < HDIM; ++hd) VsT[hd * VSTR + tid] = vr.hh[hd];
  }
  __syncthreads();

  const int w = tid >> 5;
  const int lane = tid & 31;
  const int lr = lane & 15;
  const int lg = lane >> 4;
  const int Q0 = w * 32;
  const float scale = 0.125f;  // 1/sqrt(64)

  // Q fragments (A operand), 2 row-tiles x 2 K-chunks
  v16h aQ[2][2];
#pragma unroll
  for (int i = 0; i < 2; ++i)
#pragma unroll
    for (int kc = 0; kc < 2; ++kc) {
      const _Float16* qrow =
          Qg + ((size_t)(b * TSEQ + Q0 + i * 16 + lr)) * DMODEL + h * HDIM + kc * 32;
      aQ[i][kc] = load_a_frag(qrow, lg);
    }

  float mrow[2][8], lrow[2][8];
  v8f yAcc[2][4];
#pragma unroll
  for (int i = 0; i < 2; ++i) {
#pragma unroll
    for (int r = 0; r < 8; ++r) { mrow[i][r] = -1e30f; lrow[i][r] = 0.f; }
#pragma unroll
    for (int j = 0; j < 4; ++j)
#pragma unroll
      for (int r = 0; r < 8; ++r) yAcc[i][j][r] = 0.f;
  }

  const int nkt = Q0 / 16 + 2;  // causal: key tiles covering keys <= Q0+31
  for (int kt = 0; kt < nkt; ++kt) {
    const int kb = kt * 16;

    // K^T fragments (B operand; contraction over head-dim, contiguous in LDS)
    v16h bK[2];
#pragma unroll
    for (int kc = 0; kc < 2; ++kc)
      bK[kc] = load_a_frag(&Ks[(kb + lr) * HDIM + kc * 32], lg);

    // V fragments (B operand over 32 "keys"; upper 16 are padding, contiguous
    // in the transposed layout; pad stride keeps reads in-bounds)
    v16h bV[4];
#pragma unroll
    for (int j = 0; j < 4; ++j)
      bV[j] = load_a_frag(&VsT[(j * 16 + lr) * VSTR + kb], lg);

#pragma unroll
    for (int i = 0; i < 2; ++i) {
      if (kb > Q0 + i * 16 + 15) continue;  // fully masked row-tile
      v8f s;
#pragma unroll
      for (int r = 0; r < 8; ++r) s[r] = 0.f;
      s = __builtin_amdgcn_wmma_f32_16x16x32_f16(false, aQ[i][0], false, bK[0],
                                                 (short)0, s, false, false);
      s = __builtin_amdgcn_wmma_f32_16x16x32_f16(false, aQ[i][1], false, bK[1],
                                                 (short)0, s, false, false);
#pragma unroll
      for (int r = 0; r < 8; ++r) {
        const int qr = Q0 + i * 16 + r + 8 * lg;
        const int ky = kb + lr;
        float sv = s[r] * scale;
        if (ky > qr) sv = -1e30f;  // causal mask
        float tmax = sv;
        tmax = fmaxf(tmax, __shfl_xor(tmax, 1, 32));
        tmax = fmaxf(tmax, __shfl_xor(tmax, 2, 32));
        tmax = fmaxf(tmax, __shfl_xor(tmax, 4, 32));
        tmax = fmaxf(tmax, __shfl_xor(tmax, 8, 32));
        const float mnew = fmaxf(mrow[i][r], tmax);
        const float alpha = __expf(mrow[i][r] - mnew);
        const float pp = __expf(sv - mnew);
        float rsum = pp;
        rsum += __shfl_xor(rsum, 1, 32);
        rsum += __shfl_xor(rsum, 2, 32);
        rsum += __shfl_xor(rsum, 4, 32);
        rsum += __shfl_xor(rsum, 8, 32);
        lrow[i][r] = lrow[i][r] * alpha + rsum;
        mrow[i][r] = mnew;
#pragma unroll
        for (int j = 0; j < 4; ++j) yAcc[i][j][r] *= alpha;
        Ps[w][(r + 8 * lg) * 16 + lr] = (_Float16)pp;  // stage P tile
      }
      // wave-private LDS round-trip; wait + compiler barrier
      asm volatile("s_wait_dscnt 0" ::: "memory");
      // A-operand for P: K(lower 16) = one contiguous 16B chunk, upper 16 = 0
      v16h aP;
      {
        union { v16h v; uint4 u[2]; } t;
        t.u[0] = *reinterpret_cast<const uint4*>(&Ps[w][lr * 16 + (lg << 3)]);
        uint4 z; z.x = z.y = z.z = z.w = 0u;
        t.u[1] = z;
        aP = t.v;
      }
#pragma unroll
      for (int j = 0; j < 4; ++j)
        yAcc[i][j] = __builtin_amdgcn_wmma_f32_16x16x32_f16(
            false, aP, false, bV[j], (short)0, yAcc[i][j], false, false);
    }
  }

  // write y (f16) back to [M, 768] at head column offset
#pragma unroll
  for (int i = 0; i < 2; ++i)
#pragma unroll
    for (int r = 0; r < 8; ++r) {
      const int qr = Q0 + i * 16 + r + 8 * lg;
      const float inv = 1.f / lrow[i][r];
      _Float16* yrow = Yg + ((size_t)(b * TSEQ + qr)) * DMODEL + h * HDIM;
#pragma unroll
      for (int j = 0; j < 4; ++j)
        yrow[j * 16 + lr] = (_Float16)(yAcc[i][j][r] * inv);
    }
}

// ---------------------------------------------------------------------------
// Driver
// ---------------------------------------------------------------------------
extern "C" void kernel_launch(void* const* d_in, const int* in_sizes, int n_in,
                              void* d_out, int out_size, void* d_ws, size_t ws_size,
                              hipStream_t stream) {
  (void)in_sizes; (void)n_in; (void)out_size; (void)ws_size;
  const int*   idxs = (const int*)d_in[0];
  const float* lat  = (const float*)d_in[1];
  const float* lon  = (const float*)d_in[2];
  const float* sog  = (const float*)d_in[3];
  const float* cog  = (const float*)d_in[4];
  const float* pos  = (const float*)d_in[5];
  const float* ln1g = (const float*)d_in[6];
  const float* ln1b = (const float*)d_in[7];
  const float* ln2g = (const float*)d_in[8];
  const float* ln2b = (const float*)d_in[9];
  const float* Wq   = (const float*)d_in[10];
  const float* bq   = (const float*)d_in[11];
  const float* Wk   = (const float*)d_in[12];
  const float* bk   = (const float*)d_in[13];
  const float* Wv   = (const float*)d_in[14];
  const float* bv   = (const float*)d_in[15];
  const float* Wo   = (const float*)d_in[16];
  const float* bo   = (const float*)d_in[17];
  const float* W1   = (const float*)d_in[18];
  const float* b1   = (const float*)d_in[19];
  const float* W2   = (const float*)d_in[20];
  const float* b2   = (const float*)d_in[21];
  const float* lnfg = (const float*)d_in[22];
  const float* lnfb = (const float*)d_in[23];
  const float* Wh   = (const float*)d_in[24];

  // workspace layout (bytes)
  char* ws = (char*)d_ws;
  const size_t SZ_H   = (size_t)M_TOK * DMODEL * 4;  // 50,331,648
  const size_t SZ_X16 = (size_t)M_TOK * DMODEL * 2;  // 25,165,824
  float*    hbuf = (float*)ws;
  _Float16* x16  = (_Float16*)(ws + SZ_H);
  char*     big  = ws + SZ_H + SZ_X16;               // 100.7 MB shared region
  _Float16* q16  = (_Float16*)big;
  _Float16* k16  = (_Float16*)(big + SZ_X16);
  _Float16* v16  = (_Float16*)(big + 2 * SZ_X16);
  _Float16* y16  = (_Float16*)(big + 3 * SZ_X16);
  _Float16* m16  = (_Float16*)big;  // MLP hidden aliases q/k/v/y (phase-disjoint)

  const dim3 blk(256);
  const dim3 gD((DMODEL + 63) / 64, M_TOK / 128);
  const dim3 gF((DFF + 63) / 64, M_TOK / 128);
  const dim3 gH((NFULL + 63) / 64, M_TOK / 128);

  embed_kernel<<<M_TOK, blk, 0, stream>>>(idxs, lat, lon, sog, cog, pos, hbuf);

  for (int l = 0; l < NLAYER; ++l) {
    const size_t wD = (size_t)l * DMODEL * DMODEL;
    const size_t wF = (size_t)l * DMODEL * DFF;
    ln_kernel<<<M_TOK, blk, 0, stream>>>(hbuf, ln1g + l * DMODEL, ln1b + l * DMODEL, x16);
    gemm_kernel<<<gD, blk, 0, stream>>>(x16, Wq + wD, bq + l * DMODEL, q16, DMODEL, DMODEL, 0);
    gemm_kernel<<<gD, blk, 0, stream>>>(x16, Wk + wD, bk + l * DMODEL, k16, DMODEL, DMODEL, 0);
    gemm_kernel<<<gD, blk, 0, stream>>>(x16, Wv + wD, bv + l * DMODEL, v16, DMODEL, DMODEL, 0);
    attn_kernel<<<dim3(NHEAD, M_TOK / TSEQ), blk, 0, stream>>>(q16, k16, v16, y16);
    gemm_kernel<<<gD, blk, 0, stream>>>(y16, Wo + wD, bo + l * DMODEL, hbuf, DMODEL, DMODEL, 2);
    ln_kernel<<<M_TOK, blk, 0, stream>>>(hbuf, ln2g + l * DMODEL, ln2b + l * DMODEL, x16);
    gemm_kernel<<<gF, blk, 0, stream>>>(x16, W1 + wF, b1 + l * DFF, m16, DMODEL, DFF, 1);
    gemm_kernel<<<gD, blk, 0, stream>>>(m16, W2 + wF, b2 + l * DMODEL, hbuf, DFF, DMODEL, 2);
  }

  ln_kernel<<<M_TOK, blk, 0, stream>>>(hbuf, lnfg, lnfb, x16);
  gemm_kernel<<<gH, blk, 0, stream>>>(x16, Wh, nullptr, d_out, DMODEL, NFULL, 3);
}
